// ResonantMultiheadAttention_57217554317757
// MI455X (gfx1250) — matrix operand order repaired and checked by reading, not compiled, not run on hardware
//
#include <hip/hip_runtime.h>
#include <hip/hip_bf16.h>
#include <math.h>

#define DIM     1024
#define HEADS   16
#define HD      64
#define BATCH   2
#define SEQ     2048
#define BT      (BATCH*SEQ)       /* 4096 rows */
#define THRESH  0.29514f
#define SHARP   15.0f
#define LN_EPS  1e-5f
#define NORM_EPS 1e-12f

typedef __bf16 bf16;
typedef __attribute__((ext_vector_type(16))) __bf16 v16bf;
typedef __attribute__((ext_vector_type(8)))  __bf16 v8bf;
typedef __attribute__((ext_vector_type(8)))  float  v8f;
typedef __attribute__((ext_vector_type(4)))  unsigned int u32x4;
typedef __attribute__((ext_vector_type(8)))  int i32x8;
typedef __attribute__((ext_vector_type(4)))  int i32x4;

__device__ __forceinline__ v8f wmma_bf(v16bf a, v16bf b, v8f c) {
  // D = A(16x32 bf16) * B(32x16 bf16) + C(16x16 f32)
  return __builtin_amdgcn_wmma_f32_16x16x32_bf16(false, a, false, b, (short)0, c,
                                                 false, false);
}

__device__ __forceinline__ v16bf cat16(v8bf lo, v8bf hi) {
  return __builtin_shufflevector(lo, hi, 0,1,2,3,4,5,6,7,8,9,10,11,12,13,14,15);
}

// A-matrix fragment: 16-bit A 16x32, contiguous k order (B loads match).
// lane<16: row=lane, K elems {kb..kb+15}, kb=0 ; lane>=16: kb=16.
__device__ __forceinline__ v16bf load_a_frag(const bf16* rowptr, int koff, int hi) {
  const bf16* p = rowptr + koff + hi * 16;
  v8bf lo  = *(const v8bf*)(p);
  v8bf hi8 = *(const v8bf*)(p + 8);
  return cat16(lo, hi8);
}

// ---------------------------------------------------------------------------
// Kernel 1: per-row LayerNorm; emit bf16 copies of x and x_norm.
// ---------------------------------------------------------------------------
__global__ __launch_bounds__(256)
void ln_kernel(const float* __restrict__ x, const float* __restrict__ lnw,
               const float* __restrict__ lnb, bf16* __restrict__ xbf,
               bf16* __restrict__ xnbf) {
  int row = blockIdx.x;
  int tid = threadIdx.x;
  const float* xr = x + (size_t)row * DIM;
  float4 v = ((const float4*)xr)[tid];           // 256 threads * 4 = 1024
  float s  = v.x + v.y + v.z + v.w;
  float s2 = v.x*v.x + v.y*v.y + v.z*v.z + v.w*v.w;
  #pragma unroll
  for (int m = 16; m; m >>= 1) { s += __shfl_xor(s, m, 32); s2 += __shfl_xor(s2, m, 32); }
  __shared__ float red[16];
  int wid = tid >> 5, lane = tid & 31;
  if (lane == 0) { red[wid] = s; red[wid + 8] = s2; }
  __syncthreads();
  float ts = 0.f, ts2 = 0.f;
  #pragma unroll
  for (int w = 0; w < 8; ++w) { ts += red[w]; ts2 += red[w + 8]; }
  float mu  = ts * (1.0f / DIM);
  float var = ts2 * (1.0f / DIM) - mu * mu;
  float inv = rsqrtf(var + LN_EPS);
  int j = tid * 4;
  float xe[4] = {v.x, v.y, v.z, v.w};
  #pragma unroll
  for (int i = 0; i < 4; ++i) {
    float xn = (xe[i] - mu) * inv * lnw[j + i] + lnb[j + i];
    xbf [(size_t)row * DIM + j + i] = (bf16)xe[i];
    xnbf[(size_t)row * DIM + j + i] = (bf16)xn;
  }
}

// ---------------------------------------------------------------------------
// Kernel 2: convert 4 weight matrices fp32 [K][N] -> bf16 transposed [N][K]
// so WMMA B-fragments (lane=column, contiguous K) are linear 32B loads.
// ---------------------------------------------------------------------------
__global__ __launch_bounds__(256)
void cvt_w_kernel(const float* __restrict__ w0, const float* __restrict__ w1,
                  const float* __restrict__ w2, const float* __restrict__ w3,
                  bf16* __restrict__ t0, bf16* __restrict__ t1,
                  bf16* __restrict__ t2, bf16* __restrict__ t3) {
  int which = blockIdx.y;
  const float* src = which == 0 ? w0 : which == 1 ? w1 : which == 2 ? w2 : w3;
  bf16*        dst = which == 0 ? t0 : which == 1 ? t1 : which == 2 ? t2 : t3;
  int idx = blockIdx.x * 256 + threadIdx.x;      // 0 .. 1M-1
  int n = idx >> 10, k = idx & (DIM - 1);
  dst[idx] = (bf16)src[(size_t)k * DIM + n];
}

// ---------------------------------------------------------------------------
// Kernel 3: generic bf16 WMMA GEMM, C[4096x1024] = A[4096x1024] * Bt^T.
// Wave tile: 16 rows x 64 cols (= one head), K-step 32, 4 v_wmma per step.
// Epilogue modes: 0 = plain bf16 store (V)
//                 1 = per-head L2 norm + gate dot (Q / K)
//                 2 = +bias, f32 store (output projection)
// ---------------------------------------------------------------------------
__global__ __launch_bounds__(256)
void gemm_kernel(const bf16* __restrict__ A, const bf16* __restrict__ Bt,
                 bf16* __restrict__ Cbf, float* __restrict__ Cf32,
                 const float* __restrict__ gvec, float* __restrict__ gate_out,
                 const float* __restrict__ bias, int mode) {
  const int wid  = blockIdx.x * 8 + (threadIdx.x >> 5);
  const int lane = threadIdx.x & 31, l15 = lane & 15, hi = lane >> 4;
  const int mtiles = BT / 16;                    // 256
  const int ncol = wid / mtiles;                 // 0..15 : waves in a block share B
  const int mrow = wid % mtiles;

  const bf16* arow  = A  + (size_t)(mrow * 16 + l15) * DIM;
  const bf16* bbase = Bt + (size_t)(ncol * 64) * DIM;

  v8f acc[4] = {};
  for (int kc = 0; kc < DIM; kc += 32) {
    v16bf a = load_a_frag(arow, kc, hi);
    #pragma unroll
    for (int n = 0; n < 4; ++n) {
      const bf16* bp = bbase + (size_t)(n * 16 + l15) * DIM + kc + hi * 16;
      v16bf b = *(const v16bf*)bp;
      acc[n] = wmma_bf(a, b, acc[n]);
    }
  }

  if (mode == 1) {                               // l2norm + gate (Q/K)
    const int h = ncol;
    float g[4];
    #pragma unroll
    for (int n = 0; n < 4; ++n) g[n] = gvec[n * 16 + l15];
    #pragma unroll
    for (int r = 0; r < 8; ++r) {
      float ss = 0.f, dp = 0.f;
      #pragma unroll
      for (int n = 0; n < 4; ++n) { float q = acc[n][r]; ss += q * q; dp += q * g[n]; }
      // reduce across the 16-lane half that owns this row's 64 columns
      #pragma unroll
      for (int m = 1; m < 16; m <<= 1) { ss += __shfl_xor(ss, m, 32); dp += __shfl_xor(dp, m, 32); }
      float scale = 1.0f / fmaxf(sqrtf(ss), NORM_EPS);
      int row = mrow * 16 + r + hi * 8;
      if (l15 == 0) {
        int b = row >> 11, t = row & (SEQ - 1);
        gate_out[(size_t)(b * HEADS + h) * SEQ + t] = dp * scale;
      }
      #pragma unroll
      for (int n = 0; n < 4; ++n)
        Cbf[(size_t)row * DIM + h * 64 + n * 16 + l15] = (bf16)(acc[n][r] * scale);
    }
  } else if (mode == 0) {                        // plain bf16 (V)
    #pragma unroll
    for (int r = 0; r < 8; ++r) {
      int row = mrow * 16 + r + hi * 8;
      #pragma unroll
      for (int n = 0; n < 4; ++n)
        Cbf[(size_t)row * DIM + ncol * 64 + n * 16 + l15] = (bf16)acc[n][r];
    }
  } else {                                       // bias + f32 (out proj)
    float bv[4];
    #pragma unroll
    for (int n = 0; n < 4; ++n) bv[n] = bias[ncol * 64 + n * 16 + l15];
    #pragma unroll
    for (int r = 0; r < 8; ++r) {
      int row = mrow * 16 + r + hi * 8;
      #pragma unroll
      for (int n = 0; n < 4; ++n)
        Cf32[(size_t)row * DIM + ncol * 64 + n * 16 + l15] = acc[n][r] + bv[n];
    }
  }
}

// ---------------------------------------------------------------------------
// Kernel 4: resonant attention, one (b,h,128-row q tile) per block.
// K tile fetched by the Tensor Data Mover (tensor_load_to_lds, TENSORcnt);
// V tile staged transposed by all 256 threads in parallel with the DMA.
// 8 v_wmma per 32-col s step; full-sum accumulation (no softmax normalizer).
// ---------------------------------------------------------------------------
__global__ __launch_bounds__(256)
void attn_kernel(const bf16* __restrict__ q, const bf16* __restrict__ k,
                 const bf16* __restrict__ v, const float* __restrict__ gate_q,
                 const float* __restrict__ gate_k, bf16* __restrict__ out) {
  __shared__ __align__(32) bf16 kt[32 * 64];     // k tile, row-major [s][d] (TDM dest)
  __shared__ __align__(32) bf16 vt[64 * 32];     // v tile, transposed [d][s]
  __shared__ __align__(32) bf16 pbuf[8][16 * 32];// per-wave P restage

  const int qt = blockIdx.x & 15;
  const int bh = blockIdx.x >> 4;                // 0..31
  const int b = bh >> 4, h = bh & 15;
  const int wave = threadIdx.x >> 5, lane = threadIdx.x & 31;
  const int l15 = lane & 15, hi = lane >> 4;
  const int qrow0 = qt * 128 + wave * 16;        // first q row of this wave
  const size_t headbase = (size_t)(b * SEQ) * DIM + h * 64;

  // q A-fragments (K = hd = 64 -> two 32-wide fragments), invariant over s loop
  const bf16* qrow = q + headbase + (size_t)(qrow0 + l15) * DIM;
  v16bf aq[2];
  #pragma unroll
  for (int c = 0; c < 2; ++c) aq[c] = load_a_frag(qrow, c * 32, hi);

  float gqr[8];
  #pragma unroll
  for (int r = 0; r < 8; ++r) gqr[r] = gate_q[(size_t)bh * SEQ + qrow0 + r + hi * 8];
  const float* gk = gate_k + (size_t)bh * SEQ;

  const int tid = threadIdx.x;
  const int sS = tid >> 3, dof = (tid & 7) * 8;  // staging coords

  // --- Tensor DMA descriptor (D#) for the 32x64 bf16 K tile --------------
  // group1: data_size=2B, tensor_dim0=64, tensor_dim1=2048,
  //         tile_dim0=64, tile_dim1=32, tensor_dim0_stride=1024 elems.
  const unsigned kt_lds = (unsigned)(unsigned long long)(size_t)&kt[0];
  i32x8 g1;
  g1[0] = 0x00010000;                  // workgroup_mask=0, data_size=1 (2 bytes)
  g1[1] = (int)(64u << 16);            // tensor_dim0[15:0] in bits[63:48]
  g1[2] = (int)(2048u << 16);          // tensor_dim0 hi=0, tensor_dim1[15:0]
  g1[3] = (int)(64u << 16);            // tensor_dim1 hi=0, tile_dim0=64
  g1[4] = 32;                          // tile_dim1=32, tile_dim2=0
  g1[5] = DIM;                         // tensor_dim0_stride[31:0] = 1024
  g1[6] = 0;                           // stride0 hi, stride1 lo
  g1[7] = 0;
  const i32x4 gz4 = {0, 0, 0, 0};      // groups 2/3 unused (2D tensor)
  const i32x8 gz8 = {0, 0, 0, 0, 0, 0, 0, 0};

  v8f acc[4] = {};
  for (int sb = 0; sb < SEQ; sb += 32) {
    __syncthreads();                   // previous tiles fully consumed
    if (tid < 32) {                    // wave 0: launch TDM for the K tile
      unsigned long long ga =
          (unsigned long long)(size_t)(k + headbase + (size_t)sb * DIM);
      u32x4 g0;
      g0[0] = 1u;                                        // count=1 (valid D#)
      g0[1] = kt_lds;                                    // LDS byte address
      g0[2] = (unsigned)ga;                              // global_addr[31:0]
      g0[3] = (unsigned)((ga >> 32) & 0x01FFFFFFu) | 0x80000000u; // addr hi|type=2
      __builtin_amdgcn_tensor_load_to_lds(g0, g1, gz4, gz4, gz8, 0);
    }
    // all 256 threads stage V transposed while the DMA runs
    {
      const size_t srcrow = headbase + (size_t)(sb + sS) * DIM + dof;
      v8bf vv = *(const v8bf*)(v + srcrow);
      #pragma unroll
      for (int j = 0; j < 8; ++j) vt[(dof + j) * 32 + sS] = vv[j];
    }
    if (tid < 32) __builtin_amdgcn_s_wait_tensorcnt(0);  // K tile landed
    __syncthreads();

    float gkl[2];
    #pragma unroll
    for (int n = 0; n < 2; ++n) gkl[n] = gk[sb + n * 16 + l15];

    bf16* pw = pbuf[wave];
    #pragma unroll
    for (int n = 0; n < 2; ++n) {                // two 16-col score sub-tiles
      v8f sacc = {};
      #pragma unroll
      for (int c = 0; c < 2; ++c) {
        const bf16* kp = &kt[(n * 16 + l15) * 64 + c * 32 + hi * 16];
        v16bf bk = *(const v16bf*)kp;
        sacc = wmma_bf(aq[c], bk, sacc);
      }
      #pragma unroll
      for (int r = 0; r < 8; ++r) {              // sigmoid gate, restage as bf16
        float z   = (sacc[r] - THRESH) * SHARP;
        float sg  = 1.0f / (1.0f + __expf(-z));
        float mod = sg * gqr[r] * gkl[n];
        pw[(r + hi * 8) * 32 + n * 16 + l15] = (bf16)mod;
      }
    }

    // read P back in A-fragment layout (wave-private region; HW DScnt waits)
    const bf16* pr = pw + l15 * 32 + hi * 16;
    v16bf ap = cat16(*(const v8bf*)pr, *(const v8bf*)(pr + 8));
    #pragma unroll
    for (int n2 = 0; n2 < 4; ++n2) {             // accumulate P @ V
      const bf16* vp = &vt[(n2 * 16 + l15) * 32 + hi * 16];
      v16bf bv = *(const v16bf*)vp;
      acc[n2] = wmma_bf(ap, bv, acc[n2]);
    }
  }

  #pragma unroll
  for (int r = 0; r < 8; ++r) {
    int trow = qrow0 + r + hi * 8;
    #pragma unroll
    for (int n2 = 0; n2 < 4; ++n2)
      out[headbase + (size_t)trow * DIM + n2 * 16 + l15] = (bf16)acc[n2][r];
  }
}

// ---------------------------------------------------------------------------
extern "C" void kernel_launch(void* const* d_in, const int* in_sizes, int n_in,
                              void* d_out, int out_size, void* d_ws, size_t ws_size,
                              hipStream_t stream) {
  const float* x   = (const float*)d_in[0];
  const float* Wq  = (const float*)d_in[1];
  const float* Wk  = (const float*)d_in[2];
  const float* Wv  = (const float*)d_in[3];
  const float* gq  = (const float*)d_in[4];
  const float* gk  = (const float*)d_in[5];
  const float* Wo  = (const float*)d_in[6];
  const float* bo  = (const float*)d_in[7];
  const float* lnw = (const float*)d_in[8];
  const float* lnb = (const float*)d_in[9];
  float* out = (float*)d_out;

  char* ws = (char*)d_ws;
  size_t off = 0;
  auto carve = [&](size_t bytes) -> char* {
    char* p = ws + off;
    off += (bytes + 255) & ~(size_t)255;
    return p;
  };
  bf16* xbf   = (bf16*)carve((size_t)BT * DIM * 2);
  bf16* xnbf  = (bf16*)carve((size_t)BT * DIM * 2);
  bf16* wq_t  = (bf16*)carve((size_t)DIM * DIM * 2);
  bf16* wk_t  = (bf16*)carve((size_t)DIM * DIM * 2);
  bf16* wv_t  = (bf16*)carve((size_t)DIM * DIM * 2);
  bf16* wo_t  = (bf16*)carve((size_t)DIM * DIM * 2);
  bf16* qbf   = (bf16*)carve((size_t)BT * DIM * 2);
  bf16* kbf   = (bf16*)carve((size_t)BT * DIM * 2);
  bf16* vbf   = (bf16*)carve((size_t)BT * DIM * 2);
  bf16* cbf   = (bf16*)carve((size_t)BT * DIM * 2);
  float* gateq = (float*)carve((size_t)BATCH * HEADS * SEQ * 4);
  float* gatek = (float*)carve((size_t)BATCH * HEADS * SEQ * 4);

  // 1. LayerNorm + bf16 conversion
  ln_kernel<<<BT, 256, 0, stream>>>(x, lnw, lnb, xbf, xnbf);
  // 2. weight convert + transpose (Wq, Wk, Wv, Wo)
  cvt_w_kernel<<<dim3(DIM * DIM / 256, 4), 256, 0, stream>>>(Wq, Wk, Wv, Wo,
                                                             wq_t, wk_t, wv_t, wo_t);
  // 3. projections: 512 blocks x 8 waves = 4096 wave-tiles of 16x64
  const int gblocks = (BT / 16) * (DIM / 64) / 8;   // 512
  gemm_kernel<<<gblocks, 256, 0, stream>>>(xnbf, wq_t, qbf, nullptr, gq, gateq, nullptr, 1);
  gemm_kernel<<<gblocks, 256, 0, stream>>>(xnbf, wk_t, kbf, nullptr, gk, gatek, nullptr, 1);
  gemm_kernel<<<gblocks, 256, 0, stream>>>(xbf,  wv_t, vbf, nullptr, nullptr, nullptr, nullptr, 0);
  // 4. resonant attention: B*H*(T/128) = 512 blocks
  attn_kernel<<<BATCH * HEADS * (SEQ / 128), 256, 0, stream>>>(qbf, kbf, vbf,
                                                               gateq, gatek, cbf);
  // 5. output projection + bias (f32 out)
  gemm_kernel<<<gblocks, 256, 0, stream>>>(cbf, wo_t, nullptr, out, nullptr, nullptr, bo, 2);
}
